// AttentionHead_73650099192108
// MI455X (gfx1250) — matrix operand order repaired and checked by reading, not compiled, not run on hardware
//
#include <hip/hip_runtime.h>

// Problem constants (match reference)
#define E_DIM 1024
#define H_DIM 64
#define B_DIM 8
#define T_DIM 2048

typedef __attribute__((ext_vector_type(16))) __bf16 v16bf;
typedef __attribute__((ext_vector_type(8)))  __bf16 v8bf;
typedef __attribute__((ext_vector_type(8)))  float  v8f;
typedef __attribute__((ext_vector_type(4)))  float  v4f;

union AFrag { v16bf v; v8bf h[2]; };

// sched_group_barrier masks
#define SG_VALU 0x0002
#define SG_WMMA 0x0008
#define SG_VMEM_RD 0x0020

// Workspace layout (halves):
//  [0, WFRAG_HALFS) : W B-fragments, 3 proj * 32 kstep * 4 ntile * 32 lane * 16
//  q  bf16 [B][T][H]
//  k  bf16 [B][T][H]
//  vT bf16 [B][H][T]
#define WFRAG_HALFS (3 * 32 * 4 * 32 * 16)
#define QKV_HALFS   (B_DIM * T_DIM * H_DIM)

__device__ __forceinline__ v8f wmma_bf16(v16bf a, v16bf b, v8f c) {
  return __builtin_amdgcn_wmma_f32_16x16x32_bf16(
      /*neg_a=*/false, a, /*neg_b=*/false, b,
      /*c_mod=*/(short)0, c, /*reuse_a=*/false, /*reuse_b=*/false);
}

// ---------------------------------------------------------------------------
// Kernel 0: convert fp32 weights into bf16 WMMA B-fragment layout.
// B 32x16 bf16 layout used here (contiguous k): lanes 0-15 -> N=lane, K=0..15; lanes 16-31 -> N=lane-16,
// K=16..31; element i of v16bf is K (local) = (lane/16)*16 + i.
// ---------------------------------------------------------------------------
__global__ void prep_wfrag(const float* __restrict__ Wq,
                           const float* __restrict__ Wk,
                           const float* __restrict__ Wv,
                           __bf16* __restrict__ wfrag) {
  int gid = blockIdx.x * blockDim.x + threadIdx.x;
  if (gid >= 3 * 32 * 4 * 32) return;
  int lane = gid & 31;
  int n    = (gid >> 5) & 3;
  int ks   = (gid >> 7) & 31;
  int p    = gid >> 12;
  const float* W = (p == 0) ? Wq : (p == 1) ? Wk : Wv;
  int col  = n * 16 + (lane & 15);
  int krow = ks * 32 + (lane >> 4) * 16;
  __bf16* dst = wfrag + (size_t)gid * 16;
#pragma unroll
  for (int i = 0; i < 16; ++i)
    dst[i] = (__bf16)W[(size_t)(krow + i) * H_DIM + col];
}

// ---------------------------------------------------------------------------
// Kernel 1: fused q/k/v projection. One wave = 16 rows of flattened [B*T].
// A 16x32 bf16 layout used here (contiguous k): lane<16 -> row=lane, K=0..15; lane>=16 ->
// row=lane-16, K=16..31. 12 f32 accumulators: {q,k,v} x 4 N-tiles.
// sched_group_barrier pins the per-K-step pipeline to
// [A loads][cvt][12 B loads][6 wmma][12 B loads][6 wmma] so one loadcnt wait
// covers 6 back-to-back WMMAs (critical: ~1 wave/SIMD, no cross-wave hiding).
// ---------------------------------------------------------------------------
__global__ void __launch_bounds__(128)
qkv_proj(const float* __restrict__ x, const __bf16* __restrict__ wfrag,
         __bf16* __restrict__ qws, __bf16* __restrict__ kws,
         __bf16* __restrict__ vws) {
  int tid  = threadIdx.x;
  int wid  = tid >> 5;
  int lane = tid & 31;
  int l16  = lane & 15;
  int lh   = lane >> 4;
  int tile = blockIdx.x * 4 + wid;      // 16-row tile over flattened B*T
  int rowbase = tile * 16;
  const float* xrow = x + (size_t)(rowbase + l16) * E_DIM;

  v8f acc[12];
#pragma unroll
  for (int i = 0; i < 12; ++i)
    acc[i] = (v8f){0.f, 0.f, 0.f, 0.f, 0.f, 0.f, 0.f, 0.f};

#pragma unroll 1
  for (int ks = 0; ks < 32; ++ks) {
    int kb = ks * 32 + lh * 16;
    // x is streamed exactly once kernel-wide: non-temporal.
    v4f a0 = __builtin_nontemporal_load((const v4f*)(xrow + kb));
    v4f a1 = __builtin_nontemporal_load((const v4f*)(xrow + kb + 4));
    v4f a2 = __builtin_nontemporal_load((const v4f*)(xrow + kb + 8));
    v4f a3 = __builtin_nontemporal_load((const v4f*)(xrow + kb + 12));
    v16bf A;
#pragma unroll
    for (int j = 0; j < 4; ++j) {
      A[j]      = (__bf16)a0[j];
      A[4 + j]  = (__bf16)a1[j];
      A[8 + j]  = (__bf16)a2[j];
      A[12 + j] = (__bf16)a3[j];
    }
    const __bf16* wbase =
        wfrag + ((size_t)ks * 4 * 32 + lane) * 16;  // + (p*128 + n)*32*16
#pragma unroll
    for (int g = 0; g < 2; ++g) {
      v16bf Bf[6];
#pragma unroll
      for (int j = 0; j < 6; ++j) {
        int pn = g * 6 + j;                 // 0..11
        int p = pn >> 2, n = pn & 3;
        Bf[j] = *(const v16bf*)(wbase + ((size_t)p * 128 + n) * 32 * 16);
      }
#pragma unroll
      for (int j = 0; j < 6; ++j) {
        int pn = g * 6 + j;
        acc[pn] = wmma_bf16(A, Bf[j], acc[pn]);
      }
    }
    // Pin the schedule: A loads, cvts, then (12 B-loads, 6 wmma) x 2.
    __builtin_amdgcn_sched_group_barrier(SG_VMEM_RD, 4, 0);   // A: 4x b128
    __builtin_amdgcn_sched_group_barrier(SG_VALU, 8, 0);      // 8x cvt_pk
    __builtin_amdgcn_sched_group_barrier(SG_VMEM_RD, 12, 0);  // B grp0: 12 b128
    __builtin_amdgcn_sched_group_barrier(SG_WMMA, 6, 0);      // 6 wmma
    __builtin_amdgcn_sched_group_barrier(SG_VMEM_RD, 12, 0);  // B grp1: 12 b128
    __builtin_amdgcn_sched_group_barrier(SG_WMMA, 6, 0);      // 6 wmma
  }

  // C/D layout: lane<16 -> (M=r, N=lane); lane>=16 -> (M=r+8, N=lane-16).
  int b = rowbase >> 11;  // / T_DIM
#pragma unroll
  for (int n = 0; n < 4; ++n) {
#pragma unroll
    for (int r = 0; r < 8; ++r) {
      int row = rowbase + r + 8 * lh;
      int col = n * 16 + l16;
      qws[(size_t)row * H_DIM + col] = (__bf16)acc[0 * 4 + n][r];
      kws[(size_t)row * H_DIM + col] = (__bf16)acc[1 * 4 + n][r];
      int tt = row & (T_DIM - 1);
      vws[((size_t)b * H_DIM + col) * T_DIM + tt] = (__bf16)acc[2 * 4 + n][r];
    }
  }
}

// ---------------------------------------------------------------------------
// Kernel 2: causal flash attention. One wave = 16 query rows. Online softmax
// state per (row-half, vgpr) in m[8]/lsum[8]; rows spread across the 16 lanes
// of each wave half, reduced with __shfl_xor(1,2,4,8). V fragments are loaded
// before the softmax VALU section so their latency overlaps exp/shuffles.
// Query tiles are strided across blocks to balance the causal triangle.
// ---------------------------------------------------------------------------
__global__ void __launch_bounds__(128)
attn(const __bf16* __restrict__ qws, const __bf16* __restrict__ kws,
     const __bf16* __restrict__ vws, float* __restrict__ out) {
  __shared__ __align__(32) __bf16 plds[4][16][32];

  int tid  = threadIdx.x;
  int wid  = tid >> 5;
  int lane = tid & 31;
  int l16  = lane & 15;
  int lh   = lane >> 4;
  // Balanced mapping: block i handles qtiles {i, i+G, i+2G, i+3G}.
  int qt = blockIdx.x + gridDim.x * wid;
  int b  = qt >> 7;               // / (T/16 = 128)
  int qbase = (qt & 127) * 16;

  // Q A-fragments (h 0..31 and h 32..63), loaded once.
  const __bf16* qrow = qws + (size_t)(b * T_DIM + qbase + l16) * H_DIM;
  AFrag qa0, qa1;
  qa0.h[0] = *(const v8bf*)(qrow + lh * 16);
  qa0.h[1] = *(const v8bf*)(qrow + lh * 16 + 8);
  qa1.h[0] = *(const v8bf*)(qrow + 32 + lh * 16);
  qa1.h[1] = *(const v8bf*)(qrow + 32 + lh * 16 + 8);

  v8f o[4];
#pragma unroll
  for (int n = 0; n < 4; ++n)
    o[n] = (v8f){0.f, 0.f, 0.f, 0.f, 0.f, 0.f, 0.f, 0.f};
  float m[8], lsum[8];
#pragma unroll
  for (int r = 0; r < 8; ++r) { m[r] = -1.0e30f; lsum[r] = 0.f; }

  const __bf16* kbp = kws + (size_t)b * T_DIM * H_DIM;
  const __bf16* vbp = vws + (size_t)b * H_DIM * T_DIM;

  int nkb = (qbase + 47) >> 5;    // key blocks of 32, covering keys <= qbase+15
#pragma unroll 1
  for (int kb = 0; kb < nkb; ++kb) {
    int key0 = kb * 32;
    // S = Q K^T for two 16-key tiles (K-dim = H = 64 -> 2 wmma each)
    v8f s0 = (v8f){0.f, 0.f, 0.f, 0.f, 0.f, 0.f, 0.f, 0.f};
    v8f s1 = s0;
    {
      const __bf16* kp0 = kbp + (size_t)(key0 + l16) * H_DIM;
      const __bf16* kp1 = kbp + (size_t)(key0 + 16 + l16) * H_DIM;
      v16bf b00 = *(const v16bf*)(kp0 + lh * 16);
      v16bf b01 = *(const v16bf*)(kp0 + 32 + lh * 16);
      v16bf b10 = *(const v16bf*)(kp1 + lh * 16);
      v16bf b11 = *(const v16bf*)(kp1 + 32 + lh * 16);
      s0 = wmma_bf16(qa0.v, b00, s0);
      s0 = wmma_bf16(qa1.v, b01, s0);
      s1 = wmma_bf16(qa0.v, b10, s1);
      s1 = wmma_bf16(qa1.v, b11, s1);
      __builtin_prefetch(kp0 + (size_t)32 * H_DIM);  // next key block
      // Keep the 8 K b128 loads ahead of the 4 S-WMMAs.
      __builtin_amdgcn_sched_group_barrier(SG_VMEM_RD, 8, 1);
      __builtin_amdgcn_sched_group_barrier(SG_WMMA, 4, 1);
    }
    // V fragments for this key block: issue now, consume after softmax.
    v16bf Bv[4];
#pragma unroll
    for (int n = 0; n < 4; ++n) {
      Bv[n] = *(const v16bf*)(
          vbp + (size_t)(n * 16 + l16) * T_DIM + key0 + lh * 16);
    }
    // scale + causal mask (selects, no divergence)
    int key_a = key0 + l16;
    int key_b = key0 + 16 + l16;
#pragma unroll
    for (int r = 0; r < 8; ++r) {
      int row = qbase + r + 8 * lh;
      float v0 = s0[r] * 0.125f;   // 1/sqrt(64)
      float v1 = s1[r] * 0.125f;
      s0[r] = (key_a <= row) ? v0 : -1.0e30f;
      s1[r] = (key_b <= row) ? v1 : -1.0e30f;
    }
    // online softmax update
#pragma unroll
    for (int r = 0; r < 8; ++r) {
      float tm = fmaxf(s0[r], s1[r]);
      tm = fmaxf(tm, __shfl_xor(tm, 1));
      tm = fmaxf(tm, __shfl_xor(tm, 2));
      tm = fmaxf(tm, __shfl_xor(tm, 4));
      tm = fmaxf(tm, __shfl_xor(tm, 8));
      float mn    = fmaxf(m[r], tm);
      float alpha = __expf(m[r] - mn);
      m[r] = mn;
      float p0 = __expf(s0[r] - mn);
      float p1 = __expf(s1[r] - mn);
      s0[r] = p0;
      s1[r] = p1;
      float rs = p0 + p1;
      rs += __shfl_xor(rs, 1);
      rs += __shfl_xor(rs, 2);
      rs += __shfl_xor(rs, 4);
      rs += __shfl_xor(rs, 8);
      lsum[r] = lsum[r] * alpha + rs;
#pragma unroll
      for (int n = 0; n < 4; ++n) o[n][r] *= alpha;
    }
    // P: C-layout -> A-fragment via per-wave LDS tile (intra-wave, no barrier)
#pragma unroll
    for (int r = 0; r < 8; ++r) {
      plds[wid][r + 8 * lh][l16]      = (__bf16)s0[r];
      plds[wid][r + 8 * lh][16 + l16] = (__bf16)s1[r];
    }
    __builtin_amdgcn_wave_barrier();
    AFrag pa;
    pa.h[0] = *(const v8bf*)&plds[wid][l16][lh * 16];
    pa.h[1] = *(const v8bf*)&plds[wid][l16][lh * 16 + 8];
    __builtin_amdgcn_wave_barrier();
    // O += P @ V
#pragma unroll
    for (int n = 0; n < 4; ++n) o[n] = wmma_bf16(pa.v, Bv[n], o[n]);
  }

  // normalize + store fp32 output (never re-read: non-temporal)
#pragma unroll
  for (int r = 0; r < 8; ++r) {
    float inv = 1.0f / lsum[r];
    int row = qbase + r + 8 * lh;
#pragma unroll
    for (int n = 0; n < 4; ++n) {
      __builtin_nontemporal_store(
          o[n][r] * inv,
          out + (size_t)(b * T_DIM + row) * H_DIM + n * 16 + l16);
    }
  }
}

// ---------------------------------------------------------------------------
extern "C" void kernel_launch(void* const* d_in, const int* in_sizes, int n_in,
                              void* d_out, int out_size, void* d_ws,
                              size_t ws_size, hipStream_t stream) {
  (void)in_sizes; (void)n_in; (void)out_size; (void)ws_size;
  const float* x  = (const float*)d_in[0];
  const float* Wq = (const float*)d_in[1];
  const float* Wk = (const float*)d_in[2];
  const float* Wv = (const float*)d_in[3];
  float* out = (float*)d_out;

  __bf16* wfrag = (__bf16*)d_ws;
  __bf16* qws = wfrag + WFRAG_HALFS;
  __bf16* kws = qws + QKV_HALFS;
  __bf16* vws = kws + QKV_HALFS;

  prep_wfrag<<<(3 * 32 * 4 * 32 + 127) / 128, 128, 0, stream>>>(Wq, Wk, Wv,
                                                                wfrag);
  qkv_proj<<<(B_DIM * T_DIM / 16) / 4, 128, 0, stream>>>(x, wfrag, qws, kws,
                                                         vws);
  attn<<<(B_DIM * T_DIM / 16) / 4, 128, 0, stream>>>(qws, kws, vws, out);
}
